// GATv1_1571958030452
// MI455X (gfx1250) — matrix operand order repaired and checked by reading, not compiled, not run on hardware
//
#include <hip/hip_runtime.h>
#include <hip/hip_bf16.h>

typedef __attribute__((ext_vector_type(16))) _Float16 v16h;
typedef __attribute__((ext_vector_type(8)))  float    v8f;
typedef __attribute__((ext_vector_type(4)))  int      v4i;

#define BM 64
#define BN 64
#define BK 64
#define LDP (BK + 4)   // row stride in floats: keeps 16B alignment, staggers banks

// ---- async global->LDS copy (CDNA5, ASYNCcnt) with safe fallback ----------
#if defined(__has_builtin)
#if __has_builtin(__builtin_amdgcn_global_load_async_to_lds_b128)
#define HAVE_ASYNC_LDS 1
#endif
#endif

#ifdef HAVE_ASYNC_LDS
#define ASYNC_CP16(ldsdst, gsrc)                                              \
  __builtin_amdgcn_global_load_async_to_lds_b128(                             \
      (__attribute__((address_space(1))) v4i*)(void*)(gsrc),                  \
      (__attribute__((address_space(3))) v4i*)(void*)(ldsdst), 0, 0)
#if __has_builtin(__builtin_amdgcn_s_wait_asynccnt)
#define ASYNC_WAIT_N(n) __builtin_amdgcn_s_wait_asynccnt(n)
#else
#define ASYNC_WAIT_N(n) asm volatile("s_wait_asynccnt %0" ::"i"(n) : "memory")
#endif
#else
#define ASYNC_CP16(ldsdst, gsrc) *(float4*)(ldsdst) = *(const float4*)(gsrc)
#define ASYNC_WAIT_N(n) do {} while (0)
#endif

// ---------------------------------------------------------------------------
// GEMM: C[M,N] = A[M,K] @ Bt[N,K]^T.  A f32 row-major, Bt f32 N-major
// (pre-transposed weights).  f16 WMMA, f32 accumulate.
// 256 threads = 8 waves (4x2); each wave owns 16x32 -> 2 wmma per k-step.
// K-tiles staged f32 in LDS by async b128 copies, DOUBLE-BUFFERED: tile i+1's
// DMA overlaps tile i's WMMA; s_wait_asynccnt 8 relies on in-order async
// completion (ISA 08 section 4.1) so only the older tile's copies must land.
// ---------------------------------------------------------------------------
__global__ __launch_bounds__(256) void gemm_wmma(
    const float* __restrict__ A, const float* __restrict__ Bt,
    float* __restrict__ C, int M, int K, int N)
{
  __shared__ __align__(16) float As[2][BM][LDP];
  __shared__ __align__(16) float Bs[2][BN][LDP];

  const int tid  = threadIdx.x;
  const int lane = tid & 31;
  const int wave = tid >> 5;        // 0..7
  const int wm   = wave >> 1;       // 0..3 row subtile
  const int wn   = wave & 1;        // 0..1 col group (32 cols each)
  const int half = lane >> 4;
  const int mr   = lane & 15;

  const int blockRow = blockIdx.x * BM;
  const int blockCol = blockIdx.y * BN;

  v8f acc0 = {}, acc1 = {};

  // issue one K-tile's async copies (8 x b128 per thread => ASYNCcnt += 8)
  auto issue_tile = [&](int buf, int k0) {
    #pragma unroll
    for (int t = 0; t < 4; ++t) {
      int q  = tid + t * 256;
      int r  = q >> 4;              // 0..63
      int c4 = (q & 15) << 2;       // 0,4,..,60
      int gr = blockRow + r;
      if (gr >= M) gr = M - 1;      // branchless clamp: C-row r only uses A-row r
      ASYNC_CP16(&As[buf][r][c4], A + (size_t)gr * K + k0 + c4);
    }
    #pragma unroll
    for (int t = 0; t < 4; ++t) {
      int q  = tid + t * 256;
      int n  = q >> 4;
      int c4 = (q & 15) << 2;
      ASYNC_CP16(&Bs[buf][n][c4], Bt + (size_t)(blockCol + n) * K + k0 + c4);
    }
  };

  const int ntiles = K / BK;
  issue_tile(0, 0);

  for (int i = 0; i < ntiles; ++i) {
    if (i + 1 < ntiles) {
      issue_tile((i + 1) & 1, (i + 1) * BK);   // prefetch next tile's DMA
      ASYNC_WAIT_N(8);                          // older tile done; 8 still in flight
    } else {
      ASYNC_WAIT_N(0);
    }
    __syncthreads();

    const int b = i & 1;
    #pragma unroll
    for (int ks = 0; ks < BK; ks += 32) {
      // A fragment: row (wm*16+mr), K {h*16+0..15} (contiguous, matches B)
      v16h a;
      const float* arow = &As[b][wm * 16 + mr][ks];
      #pragma unroll
      for (int j = 0; j < 4; ++j) {
        int kb  = half * 16 + 2 * j;
        int kb2 = half * 16 + 8 + 2 * j;
        a[2 * j]         = (_Float16)arow[kb];
        a[2 * j + 1]     = (_Float16)arow[kb + 1];
        a[8 + 2 * j]     = (_Float16)arow[kb2];
        a[8 + 2 * j + 1] = (_Float16)arow[kb2 + 1];
      }
      // two 16-col subtiles share the A fragment
      {
        v16h bb;
        const float* brow = &Bs[b][wn * 32 + mr][ks + half * 16];
        #pragma unroll
        for (int j = 0; j < 16; ++j) bb[j] = (_Float16)brow[j];
        acc0 = __builtin_amdgcn_wmma_f32_16x16x32_f16(false, a, false, bb,
                                                      (short)0, acc0, false, false);
      }
      {
        v16h bb;
        const float* brow = &Bs[b][wn * 32 + 16 + mr][ks + half * 16];
        #pragma unroll
        for (int j = 0; j < 16; ++j) bb[j] = (_Float16)brow[j];
        acc1 = __builtin_amdgcn_wmma_f32_16x16x32_f16(false, a, false, bb,
                                                      (short)0, acc1, false, false);
      }
    }
    __syncthreads();   // closes all reads of buffer b before it is re-written
  }

  // ---- store: lane holds rows (half*8 + r) of its 16x16 tiles ----
  const int col0 = blockCol + wn * 32 + mr;
  #pragma unroll
  for (int r = 0; r < 8; ++r) {
    int row = blockRow + wm * 16 + half * 8 + r;
    if (row < M) {
      C[(size_t)row * N + col0]      = acc0[r];
      C[(size_t)row * N + col0 + 16] = acc1[r];
    }
  }
}

// W[K][N] row-major -> Wt[N][K]
__global__ void transpose_w(const float* __restrict__ W, float* __restrict__ Wt,
                            int K, int N)
{
  int i = blockIdx.x * blockDim.x + threadIdx.x;
  if (i < K * N) {
    int k = i / N, n = i % N;
    Wt[(size_t)n * K + k] = W[i];
  }
}

// ---------------------------------------------------------------------------
// Per-node attention dots: asrc[i] = h[i,:]·att_s, adst[i] = h[i,:]·att_d
// ---------------------------------------------------------------------------
__global__ void attn_dots(const float* __restrict__ h,
                          const float* __restrict__ att_s,
                          const float* __restrict__ att_d,
                          float* __restrict__ asrc, float* __restrict__ adst,
                          int Nn, int F)
{
  int w    = (int)((blockIdx.x * (long long)blockDim.x + threadIdx.x) >> 5);
  int lane = threadIdx.x & 31;
  if (w >= Nn) return;
  float s = 0.f, d = 0.f;
  const float* hp = h + (size_t)w * F;
  for (int f = lane; f < F; f += 32) {
    float v = hp[f];
    s += v * att_s[f];
    d += v * att_d[f];
  }
  #pragma unroll
  for (int off = 16; off > 0; off >>= 1) {
    s += __shfl_down(s, off, 32);
    d += __shfl_down(d, off, 32);
  }
  if (lane == 0) { asrc[w] = s; adst[w] = d; }
}

__global__ void fill_f32(float* __restrict__ p, float v, size_t n) {
  size_t i = blockIdx.x * (size_t)blockDim.x + threadIdx.x;
  if (i < n) p[i] = v;
}

// float atomic max via sign-split int/uint trick (init must be very negative)
__device__ __forceinline__ void atomicMaxF32(float* addr, float val) {
  int bits = __float_as_int(val);
  if (bits >= 0) atomicMax((int*)addr, bits);
  else           atomicMin((unsigned int*)addr, (unsigned int)bits);
}

__global__ void edge_logits(const int* __restrict__ src, const int* __restrict__ dst,
                            const float* __restrict__ asrc, const float* __restrict__ adst,
                            float* __restrict__ ebuf, float* __restrict__ mbuf,
                            long long E)
{
  long long e = blockIdx.x * (long long)blockDim.x + threadIdx.x;
  if (e >= E) return;
  int s = src[e], d = dst[e];
  float v = asrc[s] + adst[d];
  v = (v > 0.f) ? v : 0.2f * v;      // NEG_SLOPE = 0.2
  ebuf[e] = v;
  atomicMaxF32(&mbuf[d], v);
}

__global__ void edge_expsum(const int* __restrict__ dst,
                            float* __restrict__ ebuf,
                            const float* __restrict__ mbuf, float* __restrict__ sbuf,
                            long long E)
{
  long long e = blockIdx.x * (long long)blockDim.x + threadIdx.x;
  if (e >= E) return;
  int d = dst[e];
  float v = __expf(ebuf[e] - mbuf[d]);
  ebuf[e] = v;
  atomicAdd(&sbuf[d], v);
}

// out[dst,:] += (ex/sum[dst]) * h[src,:] -- one wave per edge, L2-resident
__global__ void edge_scatter(const int* __restrict__ src, const int* __restrict__ dst,
                             const float* __restrict__ ebuf, const float* __restrict__ sbuf,
                             const float* __restrict__ h, float* __restrict__ out,
                             long long E, int F)
{
  long long w = (blockIdx.x * (long long)blockDim.x + threadIdx.x) >> 5;
  int lane = threadIdx.x & 31;
  if (w >= E) return;
  int s = src[w], d = dst[w];
  float alpha = ebuf[w] / sbuf[d];
  const float* hp = h + (size_t)s * F;
  float* op = out + (size_t)d * F;
  for (int f = lane; f < F; f += 32)
    atomicAdd(&op[f], alpha * hp[f]);
}

__global__ void bias_relu(float* __restrict__ x, const float* __restrict__ b,
                          size_t n, int F)
{
  size_t i = blockIdx.x * (size_t)blockDim.x + threadIdx.x;
  if (i >= n) return;
  float v = x[i] + b[(int)(i % F)];
  x[i] = v > 0.f ? v : 0.f;
}

// out[i,:] = log_softmax(agg[i,:] + b)  -- one wave per node, F=64 (2/lane)
__global__ void bias_log_softmax64(float* __restrict__ io, const float* __restrict__ b,
                                   int Nn)
{
  int w    = (int)((blockIdx.x * (long long)blockDim.x + threadIdx.x) >> 5);
  int lane = threadIdx.x & 31;
  if (w >= Nn) return;
  float* row = io + (size_t)w * 64;
  float v0 = row[lane]      + b[lane];
  float v1 = row[lane + 32] + b[lane + 32];
  float mx = fmaxf(v0, v1);
  #pragma unroll
  for (int off = 16; off > 0; off >>= 1) mx = fmaxf(mx, __shfl_xor(mx, off, 32));
  float se = __expf(v0 - mx) + __expf(v1 - mx);
  #pragma unroll
  for (int off = 16; off > 0; off >>= 1) se += __shfl_xor(se, off, 32);
  float ls = mx + __logf(se);
  row[lane]      = v0 - ls;
  row[lane + 32] = v1 - ls;
}

// ---------------------------------------------------------------------------
extern "C" void kernel_launch(void* const* d_in, const int* in_sizes, int n_in,
                              void* d_out, int out_size, void* d_ws, size_t ws_size,
                              hipStream_t stream) {
  const float* x     = (const float*)d_in[0];
  const int*   ei    = (const int*)d_in[1];
  const float* W1    = (const float*)d_in[2];
  const float* at_s1 = (const float*)d_in[3];
  const float* at_d1 = (const float*)d_in[4];
  const float* b1    = (const float*)d_in[5];
  const float* W2    = (const float*)d_in[6];
  const float* at_s2 = (const float*)d_in[7];
  const float* at_d2 = (const float*)d_in[8];
  const float* b2    = (const float*)d_in[9];

  const int IN = 256, HID = 128, OUT = 64;
  const int Nn = in_sizes[0] / IN;
  const long long E = in_sizes[1] / 2;
  const int* src = ei;
  const int* dst = ei + E;

  float* ws   = (float*)d_ws;
  float* h    = ws;                              // N*HID (layer2 reuses: N*OUT)
  float* agg  = h    + (size_t)Nn * HID;         // N*HID (becomes layer-2 input)
  float* asrc = agg  + (size_t)Nn * HID;         // N
  float* adst = asrc + Nn;                       // N
  float* mbuf = adst + Nn;                       // N
  float* sbuf = mbuf + Nn;                       // N
  float* ebuf = sbuf + Nn;                       // E
  float* W1t  = ebuf + E;                        // HID*IN
  float* W2t  = W1t + (size_t)HID * IN;          // OUT*HID
  float* out  = (float*)d_out;

  const int T = 256;
  const int nodeWaveBlocks = (int)(((long long)Nn * 32 + T - 1) / T);
  const int edgeBlocks     = (int)((E + T - 1) / T);
  const int edgeWaveBlocks = (int)((E * 32 + T - 1) / T);

  // pre-transpose weights (tiny) so GEMM B-tiles are contiguous-K async copies
  transpose_w<<<(IN * HID + T - 1) / T, T, 0, stream>>>(W1, W1t, IN, HID);
  transpose_w<<<(HID * OUT + T - 1) / T, T, 0, stream>>>(W2, W2t, HID, OUT);

  // ===================== Layer 1 =====================
  {
    dim3 g((Nn + BM - 1) / BM, HID / BN);
    gemm_wmma<<<g, T, 0, stream>>>(x, W1t, h, Nn, IN, HID);
  }
  attn_dots<<<nodeWaveBlocks, T, 0, stream>>>(h, at_s1, at_d1, asrc, adst, Nn, HID);
  fill_f32<<<(Nn + T - 1) / T, T, 0, stream>>>(mbuf, -3.0e38f, (size_t)Nn);
  fill_f32<<<(Nn + T - 1) / T, T, 0, stream>>>(sbuf, 0.0f, (size_t)Nn);
  edge_logits<<<edgeBlocks, T, 0, stream>>>(src, dst, asrc, adst, ebuf, mbuf, E);
  edge_expsum<<<edgeBlocks, T, 0, stream>>>(dst, ebuf, mbuf, sbuf, E);
  fill_f32<<<(int)(((size_t)Nn * HID + T - 1) / T), T, 0, stream>>>(agg, 0.0f, (size_t)Nn * HID);
  edge_scatter<<<edgeWaveBlocks, T, 0, stream>>>(src, dst, ebuf, sbuf, h, agg, E, HID);
  bias_relu<<<(int)(((size_t)Nn * HID + T - 1) / T), T, 0, stream>>>(agg, b1, (size_t)Nn * HID, HID);

  // ===================== Layer 2 =====================
  {
    dim3 g((Nn + BM - 1) / BM, OUT / BN);
    gemm_wmma<<<g, T, 0, stream>>>(agg, W2t, h, Nn, HID, OUT);
  }
  attn_dots<<<nodeWaveBlocks, T, 0, stream>>>(h, at_s2, at_d2, asrc, adst, Nn, OUT);
  fill_f32<<<(Nn + T - 1) / T, T, 0, stream>>>(mbuf, -3.0e38f, (size_t)Nn);
  fill_f32<<<(Nn + T - 1) / T, T, 0, stream>>>(sbuf, 0.0f, (size_t)Nn);
  edge_logits<<<edgeBlocks, T, 0, stream>>>(src, dst, asrc, adst, ebuf, mbuf, E);
  edge_expsum<<<edgeBlocks, T, 0, stream>>>(dst, ebuf, mbuf, sbuf, E);
  fill_f32<<<(int)(((size_t)Nn * OUT + T - 1) / T), T, 0, stream>>>(out, 0.0f, (size_t)Nn * OUT);
  edge_scatter<<<edgeWaveBlocks, T, 0, stream>>>(src, dst, ebuf, sbuf, h, out, E, OUT);
  bias_log_softmax64<<<nodeWaveBlocks, T, 0, stream>>>(out, b2, Nn);
}
